// RNN_28621662060981
// MI455X (gfx1250) — matrix operand order repaired and checked by reading, not compiled, not run on hardware
//
#include <hip/hip_runtime.h>

#define T_STEPS 256
#define BATCH   128
#define EMBD    2048
#define HID     2048
#define FC1N    256
#define NCLS    3
#define MROWS   (T_STEPS * BATCH)   // 32768

typedef __attribute__((ext_vector_type(16))) __bf16 v16bf;
typedef __attribute__((ext_vector_type(8)))  __bf16 v8bf;
typedef __attribute__((ext_vector_type(8)))  float  v8f;
typedef __attribute__((ext_vector_type(4)))  int    v4i;

// ---------- bf16 helpers ----------
__device__ __forceinline__ __bf16 f2bf(float f) {
  union { float f; unsigned u; } a; a.f = f;
  unsigned r = a.u + 0x7FFFu + ((a.u >> 16) & 1u);
  union { unsigned short s; __bf16 b; } o; o.s = (unsigned short)(r >> 16);
  return o.b;
}
__device__ __forceinline__ float bf2f(__bf16 b) {
  union { __bf16 b; unsigned short s; } i; i.b = b;
  union { unsigned u; float f; } o; o.u = ((unsigned)i.s) << 16;
  return o.f;
}

// ---------- async global->LDS copy (CDNA5), with safe fallback ----------
// Builtin signature (from toolchain diagnostic): param0 = v4i in AS1 (global),
// param1 = LDS pointer, then imm offset + imm cpol.
typedef __attribute__((address_space(1))) v4i* gas_v4i_p;
typedef __attribute__((address_space(3))) v4i* las_v4i_p;

__device__ __forceinline__ void async_cp16(const void* g, void* l) {
#if __has_builtin(__builtin_amdgcn_global_load_async_to_lds_b128)
  void* gnc = (void*)g;   // strip const, keep generic AS
  __builtin_amdgcn_global_load_async_to_lds_b128(
      (gas_v4i_p)gnc, (las_v4i_p)l, 0, 0);
#else
  *(v8bf*)l = *(const v8bf*)g;
#endif
}
__device__ __forceinline__ void wait_async_le4() {
#if __has_builtin(__builtin_amdgcn_s_wait_asynccnt)
  __builtin_amdgcn_s_wait_asynccnt(4);
#else
  asm volatile("s_wait_asynccnt 4" ::: "memory");
#endif
}
__device__ __forceinline__ void wait_async_le0() {
#if __has_builtin(__builtin_amdgcn_s_wait_asynccnt)
  __builtin_amdgcn_s_wait_asynccnt(0);
#else
  asm volatile("s_wait_asynccnt 0" ::: "memory");
#endif
}

// ---------- WMMA fragment loads ----------
// A 16x32 bf16, lane=(m=lane%16, half=lane/16):
//   elems 0..7  -> k = k0 + half*8 + i ; elems 8..15 -> k = k0 + 16 + half*8 + (i-8)
__device__ __forceinline__ v16bf load_fragA(const __bf16* __restrict__ row, int k0, int half) {
  union { v16bf v; v8bf h[2]; } u;
  u.h[0] = *(const v8bf*)(row + k0 + half * 8);
  u.h[1] = *(const v8bf*)(row + k0 + 16 + half * 8);
  return u.v;
}
// B 32x16 bf16 (col n = lane%16): elems 0..7 -> k = k0 + half*8 + i ; elems 8..15 -> k = k0 + 16 + half*8 + (i-8)
__device__ __forceinline__ v16bf load_fragB(const __bf16* __restrict__ row, int k0, int half) {
  union { v16bf v; v8bf h[2]; } u;
  const __bf16* p = row + k0 + half * 8;
  u.h[0] = *(const v8bf*)(p);
  u.h[1] = *(const v8bf*)(p + 16);
  return u.v;
}
// Same fragments out of an LDS-staged row (byte pointer to row start, 2B/elem):
__device__ __forceinline__ v16bf lds_fragA(const char* p, int half) {
  union { v16bf v; v8bf h[2]; } u;
  u.h[0] = *(const v8bf*)(p + half * 16);
  u.h[1] = *(const v8bf*)(p + 32 + half * 16);
  return u.v;
}
__device__ __forceinline__ v16bf lds_fragB(const char* p, int half) {
  union { v16bf v; v8bf h[2]; } u;
  u.h[0] = *(const v8bf*)(p + half * 16);
  u.h[1] = *(const v8bf*)(p + 32 + half * 16);
  return u.v;
}

#define WMMA_BF16(a, b, c) \
  __builtin_amdgcn_wmma_f32_16x16x32_bf16(false, (a), false, (b), (short)0, (c), false, false)

// ---------- prep kernels ----------
__global__ void f32_to_bf16_k(const float* __restrict__ src, __bf16* __restrict__ dst, int n) {
  int i = blockIdx.x * blockDim.x + threadIdx.x;
  int stride = gridDim.x * blockDim.x;
  for (; i < n; i += stride) dst[i] = f2bf(src[i]);
}

__global__ void zero_bf16_k(__bf16* __restrict__ p, int n) {
  int i = blockIdx.x * blockDim.x + threadIdx.x;
  if (i < n) p[i] = f2bf(0.0f);
}

__global__ void gather_emb_k(const int* __restrict__ x, const float* __restrict__ emb,
                             __bf16* __restrict__ A) {
  const int row = blockIdx.x;            // 0..MROWS-1, flattened t*B+b
  const int idx = x[row];
  const float*  src = emb + (size_t)idx * EMBD;
  __bf16*       dst = A   + (size_t)row * EMBD;
  for (int i = threadIdx.x; i < EMBD; i += blockDim.x) dst[i] = f2bf(src[i]);
}

// ---------- big GEMM: xproj = Xe(bf16) @ W_ih^T + b_ih + b_hh ----------
// WG tile 128x128 (8 waves of 32Mx64N). K staged through LDS in 32-wide slabs,
// double-buffered, filled with async global->LDS copies (ASYNCcnt).
// LDS row pitch 80B (64B data + 16B pad): bank stride 20 -> conflict-free frags.
#define KSTEP 32
#define PITCH 80
#define STAGE (128 * PITCH)     // 10240 B per operand per stage

__global__ void __launch_bounds__(256)
gemm_xproj_k(const __bf16* __restrict__ A, const __bf16* __restrict__ W,
             const float* __restrict__ b_ih, const float* __restrict__ b_hh,
             float* __restrict__ C) {
  __shared__ __attribute__((aligned(16))) char smem[4 * STAGE];   // A0,A1,B0,B1 = 40KB

  const int tid  = threadIdx.x;
  const int lane = tid & 31;
  const int wave = tid >> 5;
  const int wm   = wave & 3;            // 4 blocks of 32 rows
  const int wn   = wave >> 2;           // 2 blocks of 64 cols
  const int l16  = lane & 15;
  const int half = lane >> 4;
  const int Mb = blockIdx.y * 128;
  const int Nb = blockIdx.x * 128;

  // Copy-chunk assignment: 512 chunks of 16B per operand tile, 2 per thread.
  const int c0 = tid, c1 = tid + 256;
  const int r0 = c0 >> 2, q0 = (c0 & 3) * 16;       // row, byte-in-row
  const int r1 = c1 >> 2, q1 = (c1 & 3) * 16;

  const char* Abase = (const char*)(A + (size_t)Mb * EMBD);
  const char* Wbase = (const char*)(W + (size_t)Nb * EMBD);

  const v8f vzero = {0.f, 0.f, 0.f, 0.f, 0.f, 0.f, 0.f, 0.f};
  v8f acc[2][4];
#pragma unroll
  for (int mi = 0; mi < 2; ++mi)
#pragma unroll
    for (int ni = 0; ni < 4; ++ni) acc[mi][ni] = vzero;

  const int NK = EMBD / KSTEP;          // 64 stages

  // prefetch(stage ks into buffer s): 4 async 16B copies per thread
#define PREFETCH(ks, s)                                                          \
  do {                                                                           \
    const size_t kb = (size_t)(ks) * (KSTEP * 2);                                \
    const int sa = (s)*STAGE, sb = 2 * STAGE + (s)*STAGE;                        \
    async_cp16(Abase + (size_t)r0 * (EMBD * 2) + kb + q0, smem + sa + r0 * PITCH + q0); \
    async_cp16(Abase + (size_t)r1 * (EMBD * 2) + kb + q1, smem + sa + r1 * PITCH + q1); \
    async_cp16(Wbase + (size_t)r0 * (EMBD * 2) + kb + q0, smem + sb + r0 * PITCH + q0); \
    async_cp16(Wbase + (size_t)r1 * (EMBD * 2) + kb + q1, smem + sb + r1 * PITCH + q1); \
  } while (0)

  PREFETCH(0, 0);
  for (int ks = 0; ks < NK; ++ks) {
    const int s = ks & 1;
    if (ks + 1 < NK) {
      PREFETCH(ks + 1, s ^ 1);   // issue next stage first...
      wait_async_le4();          // ...then wait until only those 4 remain
    } else {
      wait_async_le0();
    }
    __syncthreads();             // all waves' stage-ks data visible in LDS

    const int sa = s * STAGE, sb = 2 * STAGE + s * STAGE;
    v16bf af[2];
    af[0] = lds_fragA(smem + sa + (wm * 32 + l16) * PITCH, half);
    af[1] = lds_fragA(smem + sa + (wm * 32 + 16 + l16) * PITCH, half);
    v16bf bfr[4];
#pragma unroll
    for (int ni = 0; ni < 4; ++ni)
      bfr[ni] = lds_fragB(smem + sb + (wn * 64 + ni * 16 + l16) * PITCH, half);
#pragma unroll
    for (int mi = 0; mi < 2; ++mi)
#pragma unroll
      for (int ni = 0; ni < 4; ++ni)
        acc[mi][ni] = WMMA_BF16(af[mi], bfr[ni], acc[mi][ni]);

    __syncthreads();             // reads done before buffer s is overwritten
  }
#undef PREFETCH

#pragma unroll
  for (int ni = 0; ni < 4; ++ni) {
    const int n = Nb + wn * 64 + ni * 16 + l16;
    const float bias = b_ih[n] + b_hh[n];
#pragma unroll
    for (int mi = 0; mi < 2; ++mi) {
      const int mbase = Mb + wm * 32 + mi * 16 + half * 8;
#pragma unroll
      for (int r = 0; r < 8; ++r)
        C[(size_t)(mbase + r) * HID + n] = acc[mi][ni][r] + bias;
    }
  }
}

// ---------- recurrent step: h_out = tanh(xp_t + h_in @ W_hh^T), bf16 in/out ----------
// Latency-bound; W_hh (8MB bf16) is L2-resident. WG tile 64x64 -> 64 WGs/step.
__global__ void __launch_bounds__(256)
gemm_step_k(const __bf16* __restrict__ Hin, const __bf16* __restrict__ W,
            const float* __restrict__ xp_t, __bf16* __restrict__ Hout) {
  const int lane = threadIdx.x & 31;
  const int wave = threadIdx.x >> 5;
  const int wm   = wave & 3;            // 4 blocks of 16 rows
  const int wn   = wave >> 2;           // 2 blocks of 32 cols
  const int l16  = lane & 15;
  const int half = lane >> 4;
  const int Mb = blockIdx.y * 64 + wm * 16;
  const int Nb = blockIdx.x * 64 + wn * 32;

  const __bf16* ar  = Hin + (size_t)(Mb + l16) * HID;
  const __bf16* br0 = W + (size_t)(Nb + l16) * HID;
  const __bf16* br1 = W + (size_t)(Nb + 16 + l16) * HID;

  const v8f vzero = {0.f, 0.f, 0.f, 0.f, 0.f, 0.f, 0.f, 0.f};
  v8f acc0 = vzero, acc1 = vzero;

  for (int k0 = 0; k0 < HID; k0 += 32) {
    v16bf af = load_fragA(ar, k0, half);
    v16bf b0 = load_fragB(br0, k0, half);
    v16bf b1 = load_fragB(br1, k0, half);
    acc0 = WMMA_BF16(af, b0, acc0);
    acc1 = WMMA_BF16(af, b1, acc1);
  }

#pragma unroll
  for (int ni = 0; ni < 2; ++ni) {
    const int n = Nb + ni * 16 + l16;
    const v8f acc = ni ? acc1 : acc0;
#pragma unroll
    for (int r = 0; r < 8; ++r) {
      const int m = Mb + half * 8 + r;
      const float v = tanhf(acc[r] + xp_t[(size_t)m * HID + n]);
      Hout[(size_t)m * HID + n] = f2bf(v);
    }
  }
}

// ---------- MLP head: out = relu(h @ fc1^T + b1) @ fc2^T + b2 ----------
__global__ void __launch_bounds__(FC1N)
head_k(const __bf16* __restrict__ h, const float* __restrict__ fc1_w,
       const float* __restrict__ fc1_b, const float* __restrict__ fc2_w,
       const float* __restrict__ fc2_b, float* __restrict__ out) {
  __shared__ float hrow[HID];
  __shared__ float z[FC1N];
  const int b = blockIdx.x;
  for (int i = threadIdx.x; i < HID; i += blockDim.x)
    hrow[i] = bf2f(h[(size_t)b * HID + i]);
  __syncthreads();
  const int n = threadIdx.x;
  float s = fc1_b[n];
  const float* wr = fc1_w + (size_t)n * HID;
  for (int k = 0; k < HID; ++k) s += hrow[k] * wr[k];
  z[n] = s > 0.f ? s : 0.f;
  __syncthreads();
  if (n < NCLS) {
    float o = fc2_b[n];
    const float* w2 = fc2_w + (size_t)n * FC1N;
    for (int k = 0; k < FC1N; ++k) o += z[k] * w2[k];
    out[b * NCLS + n] = o;
  }
}

// ---------- launcher ----------
extern "C" void kernel_launch(void* const* d_in, const int* in_sizes, int n_in,
                              void* d_out, int out_size, void* d_ws, size_t ws_size,
                              hipStream_t stream) {
  const int*   x     = (const int*)  d_in[0];
  const float* emb   = (const float*)d_in[1];
  const float* W_ih  = (const float*)d_in[2];
  const float* W_hh  = (const float*)d_in[3];
  const float* b_ih  = (const float*)d_in[4];
  const float* b_hh  = (const float*)d_in[5];
  const float* fc1_w = (const float*)d_in[6];
  const float* fc1_b = (const float*)d_in[7];
  const float* fc2_w = (const float*)d_in[8];
  const float* fc2_b = (const float*)d_in[9];
  float* out = (float*)d_out;

  char* ws = (char*)d_ws;
  const size_t OFF_A   = 0;                                    // Xe bf16: 32768x2048
  const size_t OFF_WIH = OFF_A   + (size_t)MROWS * EMBD * 2;   // W_ih bf16
  const size_t OFF_WHH = OFF_WIH + (size_t)HID * EMBD * 2;     // W_hh bf16
  const size_t OFF_XP  = OFF_WHH + (size_t)HID * HID * 2;      // xproj f32
  const size_t OFF_H0  = OFF_XP  + (size_t)MROWS * HID * 4;    // h ping (bf16)
  const size_t OFF_H1  = OFF_H0  + (size_t)BATCH * HID * 2;    // h pong (bf16)

  __bf16* Abuf  = (__bf16*)(ws + OFF_A);
  __bf16* Wihb  = (__bf16*)(ws + OFF_WIH);
  __bf16* Whhb  = (__bf16*)(ws + OFF_WHH);
  float*  xproj = (float*) (ws + OFF_XP);
  __bf16* hA    = (__bf16*)(ws + OFF_H0);
  __bf16* hB    = (__bf16*)(ws + OFF_H1);

  f32_to_bf16_k<<<2048, 256, 0, stream>>>(W_ih, Wihb, HID * EMBD);
  f32_to_bf16_k<<<2048, 256, 0, stream>>>(W_hh, Whhb, HID * HID);
  gather_emb_k<<<MROWS, 256, 0, stream>>>(x, emb, Abuf);
  zero_bf16_k<<<(BATCH * HID + 255) / 256, 256, 0, stream>>>(hA, BATCH * HID);

  gemm_xproj_k<<<dim3(HID / 128, MROWS / 128), 256, 0, stream>>>(
      Abuf, Wihb, b_ih, b_hh, xproj);

  __bf16* hc = hA; __bf16* hn = hB;
  for (int t = 0; t < T_STEPS; ++t) {
    gemm_step_k<<<dim3(HID / 64, BATCH / 64), 256, 0, stream>>>(
        hc, Whhb, xproj + (size_t)t * BATCH * HID, hn);
    __bf16* tmp = hc; hc = hn; hn = tmp;
  }

  head_k<<<BATCH, FC1N, 0, stream>>>(hc, fc1_w, fc1_b, fc2_w, fc2_b, out);

  (void)in_sizes; (void)n_in; (void)out_size; (void)ws_size;
}
